// EdgeUpdate_7627861918026
// MI455X (gfx1250) — compile-verified, matrix operand order checked by reading, not run on hardware
//
#include <hip/hip_runtime.h>

typedef __attribute__((ext_vector_type(16))) __bf16 v16bf;
typedef __attribute__((ext_vector_type(8)))  __bf16 v8bf;
typedef __attribute__((ext_vector_type(8)))  float  v8f;

// ---- CDNA5 async global->LDS path (probe via __has_builtin) --------------
#if defined(__HIP_DEVICE_COMPILE__) && defined(__has_builtin)
#if __has_builtin(__builtin_amdgcn_global_load_async_to_lds_b128) && \
    __has_builtin(__builtin_amdgcn_s_wait_asynccnt)
#define HAVE_ASYNC_LDS 1
#endif
#endif
#ifndef HAVE_ASYNC_LDS
#define HAVE_ASYNC_LDS 0
#endif

#if HAVE_ASYNC_LDS
// Builtin expects: (int4 addrspace(1)* src, int4 addrspace(3)* dst, imm off, imm cpol)
typedef int v4i_vs __attribute__((vector_size(16)));
typedef __attribute__((address_space(1))) v4i_vs gbl_v4i;
typedef __attribute__((address_space(3))) v4i_vs lds_v4i;
static __device__ __forceinline__ void async_b128(void* lds, const void* gmem) {
  __builtin_amdgcn_global_load_async_to_lds_b128(
      (gbl_v4i*)(unsigned long long)(__UINTPTR_TYPE__)gmem,
      (lds_v4i*)(unsigned)(__UINTPTR_TYPE__)lds, 0, 0);
}
#endif

static __device__ __forceinline__ v16bf cat8(v8bf lo, v8bf hi) {
  return __builtin_shufflevector(lo, hi, 0,1,2,3,4,5,6,7,8,9,10,11,12,13,14,15);
}

static __device__ __forceinline__ v8f bwmma(v16bf a, v16bf b, v8f c) {
  return __builtin_amdgcn_wmma_f32_16x16x32_bf16(false, a, false, b, (short)0, c,
                                                 false, false);
}

// dst[n*K + k] = bf16(src[k*N + n])   (transpose + f32->bf16 convert)
__global__ void cvtT_kernel(const float* __restrict__ src, __bf16* __restrict__ dst,
                            int K, int N) {
  int i = blockIdx.x * 256 + threadIdx.x;
  if (i >= K * N) return;
  int k = i / N, n = i - k * N;
  dst[n * K + k] = (__bf16)src[i];
}

// atomB[N,128] bf16 = bf16(atomF[N,D]) @ WlinT^T   (WlinT is [128][D] row-major bf16)
__global__ void __launch_bounds__(256) atom_proj_kernel(
    const float* __restrict__ atomF, const __bf16* __restrict__ WlinT,
    __bf16* __restrict__ atomB, int D) {
  const int lane = threadIdx.x & 31;
  const int w    = threadIdx.x >> 5;        // 0..7 : output column tile
  const int l15  = lane & 15;
  const int kb   = (lane >> 4) << 3;        // 0 or 8
  const int mrow = (blockIdx.x << 4) + l15; // A row for fragment loads
  const int col  = (w << 4) + l15;          // output feature column
  v8f c = {0.f,0.f,0.f,0.f,0.f,0.f,0.f,0.f};
  const int ksteps = D >> 5;
  for (int s = 0; s < ksteps; ++s) {
    const int K0 = (s << 5) + kb;
    const float* ap = atomF + (long long)mrow * D + K0;
    v8bf alo, ahi;
#pragma unroll
    for (int j = 0; j < 8; ++j) { alo[j] = (__bf16)ap[j]; ahi[j] = (__bf16)ap[16 + j]; }
    const __bf16* bp = WlinT + (long long)col * D + K0;
    v16bf a = cat8(alo, ahi);
    v16bf b = cat8(*(const v8bf*)bp, *(const v8bf*)(bp + 16));
    c = bwmma(a, b, c);
  }
  const int mbase = (blockIdx.x << 4) + ((lane >> 4) << 3);
#pragma unroll
  for (int i = 0; i < 8; ++i)
    atomB[(long long)(mbase + i) * 128 + col] = (__bf16)c[i];
}

// Fused edge MLP + residual + LayerNorm. Persistent blocks; weights in registers;
// gather/stage via async global->LDS copies.
__global__ void __launch_bounds__(256) edge_mlp_kernel(
    const __bf16* __restrict__ atomB, const float* __restrict__ edgeF,
    const long long* __restrict__ EI,
    const __bf16* __restrict__ W1T, const float* __restrict__ b1,
    const __bf16* __restrict__ W2T, const float* __restrict__ b2,
    const __bf16* __restrict__ W3T, const float* __restrict__ b3,
    const float* __restrict__ gamma, const float* __restrict__ beta,
    float* __restrict__ out, int E) {
  __shared__ __bf16 s_in[16][392];   // 16 edges x 384 (+8 pad)  bf16
  __shared__ __bf16 s_h[16][136];    // 16 x 128 (+8 pad)        bf16
  __shared__ float  s_x[16][132];    // 16 x 128 (+4 pad)        f32 (MLP out)
  __shared__ float  s_ef[16][132];   // 16 x 128 (+4 pad)        f32 (edgeF copy)

  const int tid  = threadIdx.x;
  const int lane = tid & 31;
  const int w    = tid >> 5;          // wave id -> feature column tile
  const int l15  = lane & 15;
  const int kb   = (lane >> 4) << 3;  // 0 or 8
  const int col  = (w << 4) + l15;    // 0..127
  const int mof  = (lane >> 4) << 3;  // C-fragment row offset (0 or 8)

  // ---- loop-invariant B fragments in registers -------------------------
  v16bf bw1[12], bw2[4], bw3[4];
#pragma unroll
  for (int s = 0; s < 12; ++s) {
    const __bf16* p = W1T + col * 384 + (s << 5) + kb;
    bw1[s] = cat8(*(const v8bf*)p, *(const v8bf*)(p + 16));
  }
#pragma unroll
  for (int s = 0; s < 4; ++s) {
    const __bf16* p2 = W2T + col * 128 + (s << 5) + kb;
    const __bf16* p3 = W3T + col * 128 + (s << 5) + kb;
    bw2[s] = cat8(*(const v8bf*)p2, *(const v8bf*)(p2 + 16));
    bw3[s] = cat8(*(const v8bf*)p3, *(const v8bf*)(p3 + 16));
  }
  const float bc1 = b1[col], bc2 = b2[col], bc3 = b3[col];

  const int e_loc = tid >> 4;   // 0..15 : edge within tile (staging/LN roles)
  const int ch    = tid & 15;   // 0..15 : 8-element chunk within row
  const int ntiles = E >> 4;

  for (int t = blockIdx.x; t < ntiles; t += gridDim.x) {
    const int ebase = t << 4;
    // ---- stage: gather scal[dst]|scal[src] (bf16) + edgeF (f32) into LDS
    {
      const int e = ebase + e_loc;
      const long long dst = EI[e];
      const long long src = EI[(long long)E + e];
      const __bf16* gd = atomB + dst * 128 + ch * 8;
      const __bf16* gs = atomB + src * 128 + ch * 8;
      const float*  fp = edgeF + (long long)e * 128 + ch * 8;
#if HAVE_ASYNC_LDS
      async_b128(&s_in[e_loc][ch * 8],       gd);
      async_b128(&s_in[e_loc][128 + ch * 8], gs);
      async_b128(&s_ef[e_loc][ch * 8],       fp);
      async_b128(&s_ef[e_loc][ch * 8 + 4],   fp + 4);
      __builtin_amdgcn_s_wait_asynccnt(0);
#else
      *(v8bf*)&s_in[e_loc][ch * 8]       = *(const v8bf*)gd;
      *(v8bf*)&s_in[e_loc][128 + ch * 8] = *(const v8bf*)gs;
      *(float4*)&s_ef[e_loc][ch * 8]     = *(const float4*)fp;
      *(float4*)&s_ef[e_loc][ch * 8 + 4] = *(const float4*)(fp + 4);
#endif
    }
    __syncthreads();
    // convert staged edge features (LDS f32) -> bf16 third block of s_in
    {
      float4 f0 = *(const float4*)&s_ef[e_loc][ch * 8];
      float4 f1 = *(const float4*)&s_ef[e_loc][ch * 8 + 4];
      v8bf fe;
      fe[0] = (__bf16)f0.x; fe[1] = (__bf16)f0.y;
      fe[2] = (__bf16)f0.z; fe[3] = (__bf16)f0.w;
      fe[4] = (__bf16)f1.x; fe[5] = (__bf16)f1.y;
      fe[6] = (__bf16)f1.z; fe[7] = (__bf16)f1.w;
      *(v8bf*)&s_in[e_loc][256 + ch * 8] = fe;
    }
    __syncthreads();

    // ---- GEMM1: [16,384] @ W1 -> relu -> h1 ----------------------------
    v8f c1 = {0.f,0.f,0.f,0.f,0.f,0.f,0.f,0.f};
#pragma unroll
    for (int s = 0; s < 12; ++s) {
      const __bf16* p = &s_in[l15][(s << 5) + kb];
      c1 = bwmma(cat8(*(const v8bf*)p, *(const v8bf*)(p + 16)), bw1[s], c1);
    }
#pragma unroll
    for (int i = 0; i < 8; ++i)
      s_h[mof + i][col] = (__bf16)fmaxf(c1[i] + bc1, 0.f);
    __syncthreads();

    // ---- GEMM2 ----------------------------------------------------------
    v8f c2 = {0.f,0.f,0.f,0.f,0.f,0.f,0.f,0.f};
#pragma unroll
    for (int s = 0; s < 4; ++s) {
      const __bf16* p = &s_h[l15][(s << 5) + kb];
      c2 = bwmma(cat8(*(const v8bf*)p, *(const v8bf*)(p + 16)), bw2[s], c2);
    }
    __syncthreads();   // all h1 reads complete before overwrite
#pragma unroll
    for (int i = 0; i < 8; ++i)
      s_h[mof + i][col] = (__bf16)fmaxf(c2[i] + bc2, 0.f);
    __syncthreads();

    // ---- GEMM3 -> f32 h3 ------------------------------------------------
    v8f c3 = {0.f,0.f,0.f,0.f,0.f,0.f,0.f,0.f};
#pragma unroll
    for (int s = 0; s < 4; ++s) {
      const __bf16* p = &s_h[l15][(s << 5) + kb];
      c3 = bwmma(cat8(*(const v8bf*)p, *(const v8bf*)(p + 16)), bw3[s], c3);
    }
#pragma unroll
    for (int i = 0; i < 8; ++i)
      s_x[mof + i][col] = c3[i] + bc3;
    __syncthreads();

    // ---- residual (from LDS edgeF copy) + LayerNorm --------------------
    {
      const int e = ebase + e_loc;
      float x[8], sum = 0.f, sq = 0.f;
#pragma unroll
      for (int j = 0; j < 8; ++j) {
        x[j] = s_ef[e_loc][ch * 8 + j] + s_x[e_loc][ch * 8 + j];
        sum += x[j];
        sq  += x[j] * x[j];
      }
#pragma unroll
      for (int off = 8; off >= 1; off >>= 1) {
        sum += __shfl_xor(sum, off, 16);
        sq  += __shfl_xor(sq,  off, 16);
      }
      const float mu  = sum * (1.0f / 128.0f);
      const float var = sq * (1.0f / 128.0f) - mu * mu;
      const float rs  = rsqrtf(var + 1e-5f);
      float* op = out + (long long)e * 128 + ch * 8;
      const float* gp = gamma + ch * 8;
      const float* bp = beta + ch * 8;
#pragma unroll
      for (int j = 0; j < 8; ++j)
        op[j] = (x[j] - mu) * rs * gp[j] + bp[j];
    }
    __syncthreads();
  }
}

extern "C" void kernel_launch(void* const* d_in, const int* in_sizes, int n_in,
                              void* d_out, int out_size, void* d_ws, size_t ws_size,
                              hipStream_t stream) {
  const float*     atomF = (const float*)d_in[0];
  const float*     edgeF = (const float*)d_in[1];
  const long long* EI    = (const long long*)d_in[2];
  const float*     W_lin = (const float*)d_in[3];
  const float*     W1    = (const float*)d_in[4];
  const float*     b1    = (const float*)d_in[5];
  const float*     W2    = (const float*)d_in[6];
  const float*     b2    = (const float*)d_in[7];
  const float*     W3    = (const float*)d_in[8];
  const float*     b3    = (const float*)d_in[9];
  const float*     gamma = (const float*)d_in[10];
  const float*     beta  = (const float*)d_in[11];
  float*           out   = (float*)d_out;

  const int Hh = in_sizes[5];           // 128
  const int D  = in_sizes[3] / Hh;      // 256
  const int N  = in_sizes[0] / D;       // 32000
  const int E  = in_sizes[2] / 2;       // 512000

  char* ws = (char*)d_ws;
  size_t off = 0;
  auto alloc = [&](size_t bytes) { char* p = ws + off; off = (off + bytes + 255) & ~(size_t)255; return p; };
  __bf16* atomB = (__bf16*)alloc((size_t)N * Hh * 2);
  __bf16* WlinT = (__bf16*)alloc((size_t)Hh * D * 2);
  __bf16* W1T   = (__bf16*)alloc((size_t)Hh * 3 * Hh * 2);
  __bf16* W2T   = (__bf16*)alloc((size_t)Hh * Hh * 2);
  __bf16* W3T   = (__bf16*)alloc((size_t)Hh * Hh * 2);

  // weight transpose + bf16 convert
  int n;
  n = D * Hh;      cvtT_kernel<<<(n + 255) / 256, 256, 0, stream>>>(W_lin, WlinT, D, Hh);
  n = 3 * Hh * Hh; cvtT_kernel<<<(n + 255) / 256, 256, 0, stream>>>(W1, W1T, 3 * Hh, Hh);
  n = Hh * Hh;     cvtT_kernel<<<(n + 255) / 256, 256, 0, stream>>>(W2, W2T, Hh, Hh);
  n = Hh * Hh;     cvtT_kernel<<<(n + 255) / 256, 256, 0, stream>>>(W3, W3T, Hh, Hh);

  // atom projection (bf16 WMMA)
  atom_proj_kernel<<<N / 16, 256, 0, stream>>>(atomF, WlinT, atomB, D);

  // fused gather + MLP + LayerNorm (persistent)
  edge_mlp_kernel<<<2048, 256, 0, stream>>>(atomB, edgeF, EI, W1T, b1, W2T, b2,
                                            W3T, b3, gamma, beta, out, E);
}
